// Attention_64235530879432
// MI455X (gfx1250) — matrix operand order repaired and checked by reading, not compiled, not run on hardware
//
#include <hip/hip_runtime.h>

// ---------------------------------------------------------------------------
// Types for CDNA5 WMMA (wave32): v_wmma_f32_16x16x32_bf16
// ---------------------------------------------------------------------------
typedef __bf16 bf16;
typedef __attribute__((ext_vector_type(16))) __bf16        v16bf;
typedef __attribute__((ext_vector_type(8)))  float         v8f;
typedef __attribute__((ext_vector_type(4)))  unsigned int  v4u;
typedef __attribute__((ext_vector_type(8)))  int           v8i;
typedef __attribute__((ext_vector_type(4)))  int           v4i;
typedef __attribute__((ext_vector_type(4)))  float         v4f;

union Frag16 { v16bf bf; v4u u[2]; };

// Load a 16-element bf16 fragment as two contiguous 16B chunks.
__device__ __forceinline__ v16bf ld_frag(const bf16* p0, const bf16* p1) {
  Frag16 f;
  f.u[0] = *(const v4u*)p0;
  f.u[1] = *(const v4u*)p1;
  return f.bf;
}

__device__ __forceinline__ v8f wmma_bf16(v16bf a, v16bf b, v8f c) {
  return __builtin_amdgcn_wmma_f32_16x16x32_bf16(false, a, false, b, (short)0, c,
                                                 false, false);
}

// ---------------------------------------------------------------------------
// Tensor Data Mover: 2D tile load Global -> LDS with row padding.
// D# layout per CDNA5 ISA.
// ---------------------------------------------------------------------------
#if defined(__has_builtin)
#if __has_builtin(__builtin_amdgcn_tensor_load_to_lds) && \
    __has_builtin(__builtin_amdgcn_s_wait_tensorcnt)
#define HAVE_TDM 1
#endif
#endif
#ifndef HAVE_TDM
#define HAVE_TDM 0
#endif

#if HAVE_TDM
// tile_d0 = row length (elems), tile_d1 = #rows; elements are 2 bytes (bf16).
// stride0 = row stride of the global tensor in elements.
// pad_int_code: LDS pad fires every (1<<(code+1)) DWORDs; pad_amt_code+1 DWORDs.
__device__ __forceinline__ void tdm_load_2d_bf16(
    unsigned lds_off, unsigned long long gaddr, unsigned tensor_d0,
    unsigned tensor_d1, unsigned tile_d0, unsigned tile_d1, unsigned stride0,
    unsigned pad_int_code, unsigned pad_amt_code) {
  v4u g0;
  g0.x = 1u;                                        // count=1: valid user D#
  g0.y = lds_off;                                   // lds_addr
  g0.z = (unsigned)(gaddr & 0xFFFFFFFFu);           // global_addr[31:0]
  g0.w = (unsigned)((gaddr >> 32) & 0x01FFFFFFu)    // global_addr[56:32]
         | (2u << 30);                              // type = 2 ("image")
  v8i g1;
  g1[0] = (int)((1u << 16)                          // data_size = 1 (2 bytes)
                | (1u << 20)                        // pad_enable
                | (pad_int_code << 22) | (pad_amt_code << 25));
  g1[1] = (int)((tensor_d0 & 0xFFFFu) << 16);       // tensor_dim0[15:0]
  g1[2] = (int)(((tensor_d0 >> 16) & 0xFFFFu) | ((tensor_d1 & 0xFFFFu) << 16));
  g1[3] = (int)(((tensor_d1 >> 16) & 0xFFFFu) | ((tile_d0 & 0xFFFFu) << 16));
  g1[4] = (int)(tile_d1 & 0xFFFFu);                 // tile_dim1, tile_dim2=0
  g1[5] = (int)stride0;                             // tensor_dim0_stride[31:0]
  g1[6] = 0;                                        // stride0 hi, stride1 lo
  g1[7] = 0;
  v4i z4 = {0, 0, 0, 0};
#if __clang_major__ >= 23
  v8i z8 = {0, 0, 0, 0, 0, 0, 0, 0};
  __builtin_amdgcn_tensor_load_to_lds(g0, g1, z4, z4, z8, 0);
#else
  __builtin_amdgcn_tensor_load_to_lds(g0, g1, z4, z4, 0);
#endif
}
#endif

__device__ __forceinline__ unsigned lds_offset_of(const void* p) {
  return (unsigned)(unsigned long long)(uintptr_t)p;  // flat addr low 32b = LDS offset
}

// ---------------------------------------------------------------------------
// Problem constants
// ---------------------------------------------------------------------------
#define BB   2
#define SS   2048
#define DD   4096
#define HH   32
#define KVH  8
#define HDIM 128

// ---------------------------------------------------------------------------
// f32 -> bf16 convert (8 elems/thread, vectorized)
// ---------------------------------------------------------------------------
__global__ void cvt_bf16_kernel(const float* __restrict__ src,
                                bf16* __restrict__ dst, size_t n8) {
  size_t i = (size_t)blockIdx.x * blockDim.x + threadIdx.x;
  if (i >= n8) return;
  v4f a = *(const v4f*)(src + i * 8);
  v4f b = *(const v4f*)(src + i * 8 + 4);
  bf16 o[8];
  o[0] = (bf16)a.x; o[1] = (bf16)a.y; o[2] = (bf16)a.z; o[3] = (bf16)a.w;
  o[4] = (bf16)b.x; o[5] = (bf16)b.y; o[6] = (bf16)b.z; o[7] = (bf16)b.w;
  *(v4u*)(dst + i * 8) = *(const v4u*)o;
}

// ---------------------------------------------------------------------------
// Generic bf16 x bf16 -> f32 GEMM.  C[M,N] = A[M,K] @ B[K,N]
// Block: 256 threads (8 waves). Tile 128(M) x 128(N), K step 32.
// A tile staged by TDM (row-padded), B tile staged transposed by VALU.
// ---------------------------------------------------------------------------
#define GK 32
__global__ __launch_bounds__(256) void gemm_bf16_kernel(
    const bf16* __restrict__ A, const bf16* __restrict__ Bm,
    float* __restrict__ C, int M, int K, int N) {
  __shared__ bf16 As[128][GK + 8];   // row-major A tile (+8 bf16 row pad)
  __shared__ bf16 Bt[128][GK + 8];   // transposed B tile: Bt[n][k]

  const int t    = threadIdx.x;
  const int w    = t >> 5;
  const int lane = t & 31;
  const int r    = lane & 15;
  const bool hi  = lane >= 16;

  const int mtile = blockIdx.y * 128;
  const int ntile = blockIdx.x * 128;

  v8f acc[8] = {};

  for (int kc = 0; kc < K; kc += GK) {
#if HAVE_TDM
    // A tile: 128 rows x 32 cols, row pad 8 bf16 (=4 DWORDs) every 16 DWORDs.
    if (w == 0) {
      tdm_load_2d_bf16(lds_offset_of(&As[0][0]),
                       (unsigned long long)(uintptr_t)(A + (size_t)mtile * K + kc),
                       (unsigned)K, 128u, GK, 128u, (unsigned)K,
                       /*pad_int=*/3u, /*pad_amt=*/3u);
    }
#else
    for (int i = 0; i < 8; ++i) {
      int idx = i * 256 + t;          // 0..2047 dword pairs
      int row = idx >> 4;
      int cp  = (idx & 15) << 1;
      unsigned int d =
          *(const unsigned int*)(A + (size_t)(mtile + row) * K + kc + cp);
      *(unsigned int*)&As[row][cp] = d;
    }
#endif
    // stage B transposed: 32x128 bf16, coalesced reads along N
    for (int i = 0; i < 16; ++i) {
      int idx = i * 256 + t;          // 0..4095
      int k = idx >> 7;
      int n = idx & 127;
      Bt[n][k] = Bm[(size_t)(kc + k) * N + ntile + n];
    }
    // prefetch next B k-chunk
    if (kc + GK < K)
      __builtin_prefetch(Bm + (size_t)(kc + GK) * N + ntile + (t & 127), 0, 0);
#if HAVE_TDM
    if (w == 0) __builtin_amdgcn_s_wait_tensorcnt(0);
#endif
    __syncthreads();

    // Preload all fragments, then issue the WMMA chain back-to-back.
    const bf16* ap = &As[w * 16 + r][hi ? 8 : 0];
    v16bf af = ld_frag(ap, ap + 16);
    v16bf bfg[8];
#pragma unroll
    for (int nt = 0; nt < 8; ++nt) {
      const bf16* bp = &Bt[nt * 16 + r][hi ? 8 : 0];
      bfg[nt] = ld_frag(bp, bp + 16);
    }
#pragma unroll
    for (int nt = 0; nt < 8; ++nt) acc[nt] = wmma_bf16(af, bfg[nt], acc[nt]);
    __syncthreads();
  }

  const int rowg = mtile + w * 16 + (hi ? 8 : 0);
  for (int nt = 0; nt < 8; ++nt) {
    int col = ntile + nt * 16 + r;
    for (int v = 0; v < 8; ++v)
      C[(size_t)(rowg + v) * N + col] = acc[nt][v];
  }
}

// ---------------------------------------------------------------------------
// RoPE on Q (folds 1/sqrt(HD) scale), output head-major bf16 [B,H,S,HD]
// ---------------------------------------------------------------------------
__global__ void rope_q_kernel(const float* __restrict__ q,
                              const float* __restrict__ cosb,
                              const float* __restrict__ sinb,
                              bf16* __restrict__ Qb) {
  int idx = blockIdx.x * blockDim.x + threadIdx.x;  // B*S*H*64
  int j = idx & 63;
  int h = (idx >> 6) & 31;
  int s = (idx >> 11) & 2047;
  int b = idx >> 22;
  float c  = cosb[s * 64 + j];
  float sn = sinb[s * 64 + j];
  const float* src = q + (((size_t)(b * SS + s) * HH + h) * HDIM);
  float xr = src[2 * j], xi = src[2 * j + 1];
  bf16* dst = Qb + (((size_t)(b * HH + h) * SS + s) * HDIM);
  const float qs = 0.08838834764831845f;  // 1/sqrt(128)
  dst[2 * j]     = (bf16)((xr * c - xi * sn) * qs);
  dst[2 * j + 1] = (bf16)((xr * sn + xi * c) * qs);
}

// RoPE on K, output head-major bf16 [B,KV,S,HD]
__global__ void rope_k_kernel(const float* __restrict__ k,
                              const float* __restrict__ cosb,
                              const float* __restrict__ sinb,
                              bf16* __restrict__ Kb) {
  int idx = blockIdx.x * blockDim.x + threadIdx.x;  // B*S*KV*64
  int j = idx & 63;
  int g = (idx >> 6) & 7;
  int s = (idx >> 9) & 2047;
  int b = idx >> 20;
  float c  = cosb[s * 64 + j];
  float sn = sinb[s * 64 + j];
  const float* src = k + (((size_t)(b * SS + s) * KVH + g) * HDIM);
  float xr = src[2 * j], xi = src[2 * j + 1];
  bf16* dst = Kb + (((size_t)(b * KVH + g) * SS + s) * HDIM);
  dst[2 * j]     = (bf16)(xr * c - xi * sn);
  dst[2 * j + 1] = (bf16)(xr * sn + xi * c);
}

// V: convert + relayout to [B,KV,S,HD] bf16
__global__ void conv_v_kernel(const float* __restrict__ v,
                              bf16* __restrict__ Vb) {
  int idx = blockIdx.x * blockDim.x + threadIdx.x;  // B*S*KV*HD
  int d = idx & 127;
  int g = (idx >> 7) & 7;
  int s = (idx >> 10) & 2047;
  int b = idx >> 21;
  Vb[(((size_t)(b * KVH + g) * SS + s) * HDIM) + d] =
      (bf16)v[(((size_t)(b * SS + s) * KVH + g) * HDIM) + d];
}

// ---------------------------------------------------------------------------
// Flash attention, causal, GQA (4 query heads per KV head).
// Grid: (S/128, H, B). Block: 256 threads = 8 waves; wave w owns 16 q rows.
// K tile staged by TDM; V tile transposed by VALU; P via per-wave LDS.
// ---------------------------------------------------------------------------
__global__ __launch_bounds__(256) void attn_kernel(
    const bf16* __restrict__ Qb, const bf16* __restrict__ Kb,
    const bf16* __restrict__ Vb, bf16* __restrict__ attnb) {
  __shared__ bf16 Ks[32][HDIM + 8];      // K tile, row = key (+8 bf16 row pad)
  __shared__ bf16 Vt[HDIM][32 + 8];      // V tile transposed: Vt[d][k]
  __shared__ bf16 Pw[8][16][32 + 8];     // per-wave P tile (16 q x 32 k)

  const int t    = threadIdx.x;
  const int w    = t >> 5;
  const int lane = t & 31;
  const int r    = lane & 15;
  const bool hi  = lane >= 16;

  const int qtile = blockIdx.x;
  const int h     = blockIdx.y;
  const int b     = blockIdx.z;
  const int g     = h >> 2;                    // kv head (REP = 4)
  const int qbase = qtile * 128 + w * 16;

  // Q fragments held in registers (scaled bf16), 4 K-chunks of 32 over HD=128
  v16bf qf[4];
  {
    const bf16* qrow = Qb + (((size_t)(b * HH + h) * SS + qbase + r) * HDIM);
#pragma unroll
    for (int dc = 0; dc < 4; ++dc) {
      const bf16* p = qrow + dc * 32 + (hi ? 8 : 0);
      qf[dc] = ld_frag(p, p + 16);
    }
  }

  v8f o[8] = {};
  float mrow[8], srow[8];
  for (int v = 0; v < 8; ++v) { mrow[v] = -3.0e38f; srow[v] = 0.0f; }

  const bf16* Kbase = Kb + ((size_t)(b * KVH + g) * SS) * HDIM;
  const bf16* Vbase = Vb + ((size_t)(b * KVH + g) * SS) * HDIM;

  const int kend = qtile * 128 + 128;  // causal bound (uniform per block)
  for (int kt = 0; kt < kend; kt += 32) {
    __syncthreads();  // protect Ks/Vt against previous iteration's readers
#if HAVE_TDM
    // K tile: 32 rows x 128 cols, row pad 8 bf16 (=4 DWORDs) every 64 DWORDs.
    if (w == 0) {
      tdm_load_2d_bf16(lds_offset_of(&Ks[0][0]),
                       (unsigned long long)(uintptr_t)(Kbase + (size_t)kt * HDIM),
                       HDIM, 32u, HDIM, 32u, HDIM,
                       /*pad_int=*/5u, /*pad_amt=*/3u);
    }
#else
    for (int i = 0; i < 8; ++i) {
      int idx = i * 256 + t;
      int row = idx >> 6;
      int cp  = (idx & 63) << 1;
      *(unsigned int*)&Ks[row][cp] =
          *(const unsigned int*)(Kbase + (size_t)(kt + row) * HDIM + cp);
    }
#endif
    // stage V transposed: 32x128 elements, 16 per thread
    for (int i = 0; i < 16; ++i) {
      int idx = i * 256 + t;
      int k = idx >> 7;
      int d = idx & 127;
      Vt[d][k] = Vbase[(size_t)(kt + k) * HDIM + d];
    }
    if (kt + 32 < kend)
      __builtin_prefetch(Kbase + (size_t)(kt + 32) * HDIM + t * 16, 0, 0);
#if HAVE_TDM
    if (w == 0) __builtin_amdgcn_s_wait_tensorcnt(0);
#endif
    __syncthreads();

    // scores: two 16x16 tiles (keys kt..+15, kt+16..+31), K-depth 128
    v16bf kf[8];
#pragma unroll
    for (int dc = 0; dc < 4; ++dc) {
      const bf16* kp0 = &Ks[r][dc * 32 + (hi ? 8 : 0)];
      kf[2 * dc] = ld_frag(kp0, kp0 + 16);
      const bf16* kp1 = &Ks[16 + r][dc * 32 + (hi ? 8 : 0)];
      kf[2 * dc + 1] = ld_frag(kp1, kp1 + 16);
    }
    v8f c0 = {}, c1 = {};
#pragma unroll
    for (int dc = 0; dc < 4; ++dc) {
      c0 = wmma_bf16(qf[dc], kf[2 * dc], c0);
      c1 = wmma_bf16(qf[dc], kf[2 * dc + 1], c1);
    }

    // causal mask + online softmax (rows live per half-wave: lanes 0-15 / 16-31)
#pragma unroll
    for (int v = 0; v < 8; ++v) {
      int row  = qbase + v + (hi ? 8 : 0);
      int col0 = kt + r;
      float s0 = c0[v] + (col0 > row ? -1.0e9f : 0.0f);
      float s1 = c1[v] + (col0 + 16 > row ? -1.0e9f : 0.0f);
      float mt = fmaxf(s0, s1);
      for (int off = 1; off < 16; off <<= 1)
        mt = fmaxf(mt, __shfl_xor(mt, off, 32));
      float mnew  = fmaxf(mrow[v], mt);
      float alpha = __expf(mrow[v] - mnew);
      float e0 = __expf(s0 - mnew);
      float e1 = __expf(s1 - mnew);
      float ssum = e0 + e1;
      for (int off = 1; off < 16; off <<= 1)
        ssum += __shfl_xor(ssum, off, 32);
      srow[v] = srow[v] * alpha + ssum;
      mrow[v] = mnew;
      for (int dt = 0; dt < 8; ++dt) o[dt][v] *= alpha;
      int mloc = v + (hi ? 8 : 0);
      Pw[w][mloc][r]      = (bf16)e0;
      Pw[w][mloc][r + 16] = (bf16)e1;
    }
    __syncthreads();  // P visible in A-fragment layout

    // O += P @ V   (A = P 16x32, B = V 32x16 per 16-wide d tile)
    const bf16* pp = &Pw[w][r][hi ? 8 : 0];
    v16bf pf = ld_frag(pp, pp + 16);
    v16bf vf[8];
#pragma unroll
    for (int dt = 0; dt < 8; ++dt) {
      const bf16* vp = &Vt[dt * 16 + r][hi ? 8 : 0];
      vf[dt] = ld_frag(vp, vp + 16);
    }
#pragma unroll
    for (int dt = 0; dt < 8; ++dt) o[dt] = wmma_bf16(pf, vf[dt], o[dt]);
  }

  // normalize and store bf16 [B, S, H*HD]
  float inv[8];
  for (int v = 0; v < 8; ++v) inv[v] = 1.0f / srow[v];
  for (int dt = 0; dt < 8; ++dt) {
    int d = dt * 16 + r;
    for (int v = 0; v < 8; ++v) {
      int sidx = qbase + v + (hi ? 8 : 0);
      attnb[(((size_t)(b * SS + sidx) * HH) + h) * HDIM + d] =
          (bf16)(o[dt][v] * inv[v]);
    }
  }
}

// ---------------------------------------------------------------------------
// Host-side orchestration
// ---------------------------------------------------------------------------
extern "C" void kernel_launch(void* const* d_in, const int* in_sizes, int n_in,
                              void* d_out, int out_size, void* d_ws,
                              size_t ws_size, hipStream_t stream) {
  const float* x    = (const float*)d_in[0];
  const float* cosb = (const float*)d_in[1];
  const float* sinb = (const float*)d_in[2];
  // d_in[3] = mask (computed analytically), d_in[8] = positions (arange)
  const float* wq = (const float*)d_in[4];
  const float* wk = (const float*)d_in[5];
  const float* wv = (const float*)d_in[6];
  const float* wo = (const float*)d_in[7];
  float* out = (float*)d_out;

  const size_t BS = (size_t)BB * SS;  // 4096

  char* ws = (char*)d_ws;
  size_t off = 0;
  auto take = [&](size_t bytes) {
    size_t o = off;
    off += (bytes + 255) & ~(size_t)255;
    return o;
  };
  bf16* xb    = (bf16*)(ws + take(BS * DD * 2));
  bf16* wqb   = (bf16*)(ws + take((size_t)DD * HH * HDIM * 2));
  bf16* wkb   = (bf16*)(ws + take((size_t)DD * KVH * HDIM * 2));
  bf16* wvb   = (bf16*)(ws + take((size_t)DD * KVH * HDIM * 2));
  bf16* wob   = (bf16*)(ws + take((size_t)HH * HDIM * DD * 2));
  float* qtmp = (float*)(ws + take(BS * HH * HDIM * 4));
  float* ktmp = (float*)(ws + take(BS * KVH * HDIM * 4));
  float* vtmp = (float*)(ws + take(BS * KVH * HDIM * 4));
  bf16* Qb    = (bf16*)(ws + take(BS * HH * HDIM * 2));
  bf16* Kb    = (bf16*)(ws + take(BS * KVH * HDIM * 2));
  bf16* Vb    = (bf16*)(ws + take(BS * KVH * HDIM * 2));
  bf16* attnb = (bf16*)(ws + take(BS * HH * HDIM * 2));
  (void)ws_size;

  auto cvt = [&](const float* s, bf16* d, size_t n) {
    size_t n8 = n / 8;
    cvt_bf16_kernel<<<(unsigned)((n8 + 255) / 256), 256, 0, stream>>>(s, d, n8);
  };
  cvt(x, xb, BS * DD);
  cvt(wq, wqb, (size_t)DD * HH * HDIM);
  cvt(wk, wkb, (size_t)DD * KVH * HDIM);
  cvt(wv, wvb, (size_t)DD * KVH * HDIM);
  cvt(wo, wob, (size_t)HH * HDIM * DD);

  // QKV projections
  gemm_bf16_kernel<<<dim3(HH * HDIM / 128, BS / 128), 256, 0, stream>>>(
      xb, wqb, qtmp, (int)BS, DD, HH * HDIM);
  gemm_bf16_kernel<<<dim3(KVH * HDIM / 128, BS / 128), 256, 0, stream>>>(
      xb, wkb, ktmp, (int)BS, DD, KVH * HDIM);
  gemm_bf16_kernel<<<dim3(KVH * HDIM / 128, BS / 128), 256, 0, stream>>>(
      xb, wvb, vtmp, (int)BS, DD, KVH * HDIM);

  // RoPE + relayout to head-major bf16
  rope_q_kernel<<<(unsigned)(BS * HH * 64 / 256), 256, 0, stream>>>(qtmp, cosb,
                                                                    sinb, Qb);
  rope_k_kernel<<<(unsigned)(BS * KVH * 64 / 256), 256, 0, stream>>>(ktmp, cosb,
                                                                     sinb, Kb);
  conv_v_kernel<<<(unsigned)(BS * KVH * HDIM / 256), 256, 0, stream>>>(vtmp, Vb);

  // attention
  attn_kernel<<<dim3(SS / 128, HH, BB), 256, 0, stream>>>(Qb, Kb, Vb, attnb);

  // output projection -> f32 d_out
  gemm_bf16_kernel<<<dim3(DD / 128, BS / 128), 256, 0, stream>>>(
      attnb, wob, out, (int)BS, HH * HDIM, DD);
  (void)in_sizes; (void)n_in; (void)out_size;
}
